// LossPredictor_57707180589126
// MI455X (gfx1250) — compile-verified, matrix operand order checked by reading, not run on hardware
//
#include <hip/hip_runtime.h>
#include <hip/hip_bf16.h>
#include <math.h>

typedef _Float16 v16h __attribute__((ext_vector_type(16)));
typedef _Float16 v8h  __attribute__((ext_vector_type(8)));
typedef float    v8f  __attribute__((ext_vector_type(8)));

union Frag { v16h v; v8h h[2]; };

#define B_      32
#define NEV     16
#define WIN     512
#define STEP_   256
#define NCO     257
#define EDIM    4121
#define NSAMP   32768
#define NFR     128
#define NB      128
#define MD      128
#define PDIM    33
#define INCH    4282

// stride (elements) per phase copy:
//  - multiple of 8  -> every fragment read stays 16B-aligned (ds_load_b128)
//  - XSTR*2 mod 256 == 32 -> lane bank-starts {ph*8 + (l16>=8)*4} banks form a
//    perfect permutation of the 64 LDS banks across 16 lanes (2-cycle minimum
//    for a 512B wave32 b128 request, vs 4 cycles with stride 776)
#define XSTR    784

// ---------------- f32 -> f16 conversion of x and fb ----------------
__global__ __launch_bounds__(256) void k_convert(const float* __restrict__ x,
                                                 const float* __restrict__ fb,
                                                 _Float16* __restrict__ xh,
                                                 _Float16* __restrict__ fbh) {
  int stride = gridDim.x * blockDim.x;
  int i0 = blockIdx.x * blockDim.x + threadIdx.x;
  for (int i = i0; i < B_ * NSAMP; i += stride) xh[i] = (_Float16)x[i];
  for (int i = i0; i < NB * WIN;   i += stride) fbh[i] = (_Float16)fb[i];
}

// ---------------- WMMA conv + |.| + 256-block row-sum ----------------
// grid: B_*NFR workgroups (one per (batch, 256-sample block q)), 256 threads (8 waves)
// wave w computes bands [16w,16w+16) x 256 samples with K=512, writes
// S[b][band][q] = sum_{t in block q} |conv(b,band,t)|
__global__ __launch_bounds__(256) void k_conv_wmma(const _Float16* __restrict__ xh,
                                                   const _Float16* __restrict__ fbh,
                                                   float* __restrict__ S) {
  // 8 phase-shifted copies of the 768-sample Toeplitz slice so every lane's
  // 8-half contiguous fragment read is a 16B-aligned, bank-conflict-minimal
  // ds_load_b128.
  __shared__ __align__(16) _Float16 xs[8 * XSTR];

  int wg  = blockIdx.x;
  int b   = wg >> 7;
  int q   = wg & 127;
  int tid = threadIdx.x;

  const _Float16* xrow = xh + (size_t)b * NSAMP;
  int base = q * STEP_ - STEP_;   // xpad[i] = x[b][base + i], i in [0,768)
  for (int p = 0; p < 8; ++p) {
    for (int i = tid; i < XSTR; i += 256) {
      int gx = base + i + p;
      _Float16 v = (_Float16)0.f;
      if (i < 768 && gx >= 0 && gx < NSAMP) v = xrow[gx];
      xs[p * XSTR + i] = v;
    }
  }
  __syncthreads();

  int wave = tid >> 5;
  int lane = tid & 31;
  int l16  = lane & 15;
  int half = lane >> 4;
  int ph   = l16 & 7;                         // per-lane phase -> 16B alignment
  const _Float16* xb = xs + ph * XSTR - ph;   // xb[a] == xpad[a] when a%8==ph

  // A row for this lane: fb[band = 16*wave + l16][:]
  const _Float16* Arow = fbh + (size_t)(wave * 16 + l16) * WIN;

  float bacc[8];
  #pragma unroll
  for (int r = 0; r < 8; ++r) bacc[r] = 0.f;

  for (int pass = 0; pass < 2; ++pass) {       // 2 x 8 N-tiles = 256 samples
    v8f C[8];
    #pragma unroll
    for (int n = 0; n < 8; ++n)
      #pragma unroll
      for (int r = 0; r < 8; ++r) C[n][r] = 0.f;

    int nbase = pass * 128;
    for (int kt = 0; kt < 16; ++kt) {          // K = 16 x 32
      // 16-bit A 16x32 layout (ISA 7.12.2): for this lane, dwords 0-3 hold
      // K = k0h..k0h+7 and dwords 4-7 hold K = k0h+16..k0h+23 (contiguous runs)
      int k0h = kt * 32 + half * 8;
      Frag a;
      a.h[0] = *(const v8h*)(Arow + k0h);
      a.h[1] = *(const v8h*)(Arow + k0h + 16);

      #pragma unroll
      for (int n = 0; n < 8; ++n) {
        int cb = nbase + n * 16 + l16;         // B column (sample in block)
        // B fragment: B[k][col] = xpad[col + k]; column-per-lane, K map mirrors A
        const _Float16* pB = xb + cb + k0h;
        Frag bf;
        bf.h[0] = *(const v8h*)(pB);
        bf.h[1] = *(const v8h*)(pB + 16);
        C[n] = __builtin_amdgcn_wmma_f32_16x16x32_f16(
            false, a.v, false, bf.v, (short)0, C[n], false, false);
      }
    }
    // fused epilogue: abs + per-lane partial row-sum over this pass's columns
    #pragma unroll
    for (int n = 0; n < 8; ++n)
      #pragma unroll
      for (int r = 0; r < 8; ++r) bacc[r] += fabsf(C[n][r]);
  }

  // cross-lane sum over the 16 columns held within each lane half
  #pragma unroll
  for (int r = 0; r < 8; ++r) {
    float v = bacc[r];
    v += __shfl_xor(v, 1, 32);
    v += __shfl_xor(v, 2, 32);
    v += __shfl_xor(v, 4, 32);
    v += __shfl_xor(v, 8, 32);
    bacc[r] = v;
  }
  if (l16 == 0) {
    int band = wave * 16 + half * 8;           // C row r -> band + r
    #pragma unroll
    for (int r = 0; r < 8; ++r)
      S[((size_t)b * NB + band + r) * NFR + q] = bacc[r];
  }
}

// ---------------- event contribution: evh[b][e][o] = rw[o,161:] . events[b,e,:] ----------------
__global__ __launch_bounds__(128) void k_evh(const float* __restrict__ events,
                                             const float* __restrict__ rw,
                                             float* __restrict__ evh) {
  __shared__ float ev[EDIM];
  int be = blockIdx.x;                         // b*16 + e
  const float* src = events + (size_t)be * EDIM;
  for (int i = threadIdx.x; i < EDIM; i += 128) ev[i] = src[i];
  __syncthreads();
  int o = threadIdx.x;
  const float* r = rw + (size_t)o * INCH + (NB + PDIM);
  float acc = 0.f;
  for (int c = 0; c < EDIM; ++c) acc += r[c] * ev[c];
  evh[(size_t)be * MD + o] = acc;
}

// ---------------- scatter index: last event mapping to each frame ----------------
__global__ void k_eidx(const float* __restrict__ events, int* __restrict__ eidx) {
  int b = threadIdx.x;
  if (b >= B_) return;
  for (int t = 0; t < NFR; ++t) eidx[b * NFR + t] = -1;
  for (int e = 0; e < NEV; ++e) {              // serial: last writer wins
    float tm = events[((size_t)b * NEV + e) * EDIM];
    int t = (int)(tm * (float)NFR);
    t = t < 0 ? 0 : (t > NFR - 1 ? NFR - 1 : t);
    eidx[b * NFR + t] = e;
  }
}

// ---------------- h0[b,t,o] = reduce_w . feat + reduce_b (decomposed) ----------------
__global__ __launch_bounds__(128) void k_reduce(const float* __restrict__ S,
                                                const float* __restrict__ rw,
                                                const float* __restrict__ rb,
                                                const float* __restrict__ evh,
                                                const int* __restrict__ eidx,
                                                float* __restrict__ h0) {
  __shared__ float xp[NB];
  __shared__ float pe[PDIM];
  int bt = blockIdx.x;
  int b = bt >> 7, t = bt & 127;
  int tid = threadIdx.x;
  {
    float s1 = S[((size_t)b * NB + tid) * NFR + t];
    float s0 = (t > 0) ? S[((size_t)b * NB + tid) * NFR + t - 1] : 0.f;
    xp[tid] = (s0 + s1) * (1.f / (float)WIN);
  }
  if (tid < PDIM) {
    float pos = -1.f + 2.f * (float)t / (float)(NFR - 1);
    float v;
    if (tid == 0)       v = pos;
    else if (tid <= 16) v = sinf(pos * 3.14159265358979f * (float)(1 << (tid - 1)));
    else                v = cosf(pos * 3.14159265358979f * (float)(1 << (tid - 17)));
    pe[tid] = v;
  }
  __syncthreads();
  int o = tid;
  const float* r = rw + (size_t)o * INCH;
  float acc = rb[o];
  for (int c = 0; c < NB; ++c)   acc += r[c] * xp[c];
  for (int p = 0; p < PDIM; ++p) acc += r[NB + p] * pe[p];
  int e = eidx[b * NFR + t];
  if (e >= 0) acc += evh[((size_t)b * NEV + e) * MD + o];
  h0[(size_t)bt * MD + o] = acc;
}

// ---------------- dense 128->128 + leaky relu ----------------
__global__ __launch_bounds__(128) void k_dense_lrelu(const float* __restrict__ hin,
                                                     const float* __restrict__ w,
                                                     const float* __restrict__ bias,
                                                     float* __restrict__ hout) {
  __shared__ float hs[MD];
  int row = blockIdx.x;
  int o = threadIdx.x;
  hs[o] = hin[(size_t)row * MD + o];
  __syncthreads();
  const float* wr = w + (size_t)o * MD;
  float acc = bias[o];
  for (int k = 0; k < MD; ++k) acc += wr[k] * hs[k];
  hout[(size_t)row * MD + o] = acc > 0.f ? acc : 0.2f * acc;
}

// ---------------- final dense 128->257 ----------------
__global__ __launch_bounds__(256) void k_final(const float* __restrict__ hin,
                                               const float* __restrict__ w3,
                                               const float* __restrict__ b3,
                                               float* __restrict__ out) {
  __shared__ float hs[MD];
  int row = blockIdx.x;
  int tid = threadIdx.x;
  if (tid < MD) hs[tid] = hin[(size_t)row * MD + tid];
  __syncthreads();
  for (int o = tid; o < NCO; o += 256) {
    const float* wr = w3 + (size_t)o * MD;
    float acc = b3[o];
    for (int k = 0; k < MD; ++k) acc += wr[k] * hs[k];
    out[(size_t)row * NCO + o] = acc;
  }
}

extern "C" void kernel_launch(void* const* d_in, const int* in_sizes, int n_in,
                              void* d_out, int out_size, void* d_ws, size_t ws_size,
                              hipStream_t stream) {
  const float* events = (const float*)d_in[0];
  const float* x      = (const float*)d_in[1];
  const float* fb     = (const float*)d_in[2];
  const float* rw     = (const float*)d_in[3];
  const float* rb     = (const float*)d_in[4];
  const float* w1     = (const float*)d_in[5];
  const float* b1     = (const float*)d_in[6];
  const float* w2     = (const float*)d_in[7];
  const float* b2     = (const float*)d_in[8];
  const float* w3     = (const float*)d_in[9];
  const float* b3     = (const float*)d_in[10];
  float* out = (float*)d_out;

  char* ws = (char*)d_ws;
  size_t off = 0;
  auto carve = [&](size_t bytes) {
    void* p = ws + off;
    off = (off + bytes + 255) & ~(size_t)255;
    return p;
  };
  _Float16* xh  = (_Float16*)carve((size_t)B_ * NSAMP * sizeof(_Float16));
  _Float16* fbh = (_Float16*)carve((size_t)NB * WIN * sizeof(_Float16));
  float* S   = (float*)carve((size_t)B_ * NB * NFR * sizeof(float));
  float* evh = (float*)carve((size_t)B_ * NEV * MD * sizeof(float));
  int*   eix = (int*)carve((size_t)B_ * NFR * sizeof(int));
  float* h0  = (float*)carve((size_t)B_ * NFR * MD * sizeof(float));
  float* h1  = (float*)carve((size_t)B_ * NFR * MD * sizeof(float));

  k_convert   <<<1024, 256, 0, stream>>>(x, fb, xh, fbh);
  k_conv_wmma <<<B_ * NFR, 256, 0, stream>>>(xh, fbh, S);
  k_evh       <<<B_ * NEV, 128, 0, stream>>>(events, rw, evh);
  k_eidx      <<<1, 32, 0, stream>>>(events, eix);
  k_reduce    <<<B_ * NFR, 128, 0, stream>>>(S, rw, rb, evh, eix, h0);
  k_dense_lrelu<<<B_ * NFR, 128, 0, stream>>>(h0, w1, b1, h1);
  k_dense_lrelu<<<B_ * NFR, 128, 0, stream>>>(h1, w2, b2, h0);
  k_final     <<<B_ * NFR, 256, 0, stream>>>(h0, w3, b3, out);
}
